// Model_45045617000498
// MI455X (gfx1250) — matrix operand order repaired and checked by reading, not compiled, not run on hardware
//
#include <hip/hip_runtime.h>
#include <cstdint>

typedef _Float16 half_t;
typedef __attribute__((ext_vector_type(16))) _Float16 v16h;
typedef __attribute__((ext_vector_type(8)))  _Float16 v8h;
typedef __attribute__((ext_vector_type(8)))  float    v8f;

// ---------------------------------------------------------------- utilities
__global__ void fill_u32_kernel(unsigned* p, int n, unsigned v) {
    int i = blockIdx.x * blockDim.x + threadIdx.x;
    if (i < n) p[i] = v;
}

// ------------------------------------------------- sparse submanifold convs
// contrib[p,d] = sum_c feats[in[p],c] * W[k[p],c,d]; scatter-add to out[out[p],d]
__global__ void subconv_kernel(const float* __restrict__ feats, int cin,
                               const float* __restrict__ W, int cout,
                               const int* __restrict__ iin,
                               const int* __restrict__ iout,
                               const int* __restrict__ ik,
                               int P, float* __restrict__ out) {
    int idx = blockIdx.x * blockDim.x + threadIdx.x;
    if (idx >= P * cout) return;
    int p = idx / cout, d = idx - p * cout;
    int s = iin[p], k = ik[p];
    const float* w = W + (size_t)k * cin * cout + d;
    const float* f = feats + (size_t)s * cin;
    float a = 0.f;
    for (int c = 0; c < cin; ++c) a += f[c] * w[c * cout];
    atomicAdd(out + (size_t)iout[p] * cout + d, a);
}

// ------------------------------------------------------- pooling segment_max
__device__ __forceinline__ unsigned enc_f32(float f) {
    unsigned b = __float_as_uint(f);
    return (b & 0x80000000u) ? ~b : (b | 0x80000000u);
}
__global__ void pool_max_kernel(const float* __restrict__ h3,
                                const int* __restrict__ seg,
                                unsigned* __restrict__ hp, int N) {
    int idx = blockIdx.x * blockDim.x + threadIdx.x;
    if (idx >= N * 4) return;
    int s = idx >> 2, c = idx & 3;
    atomicMax(hp + (size_t)seg[s] * 4 + c, enc_f32(h3[idx]));
}
__global__ void decode_pad_kernel(const unsigned* __restrict__ hp,
                                  float* __restrict__ h, int Np) {
    int idx = blockIdx.x * blockDim.x + threadIdx.x;
    if (idx >= Np * 16) return;
    int sp = idx >> 4, c = idx & 15;
    float v = 0.f;
    if (c < 4) {
        unsigned e = hp[sp * 4 + c];
        unsigned b = (e & 0x80000000u) ? (e & 0x7FFFFFFFu) : ~e;
        v = __uint_as_float(b);
    }
    h[idx] = v;
}

// ------------------------------------------------------------- elementwise
__global__ void bnrelu_kernel(const float* __restrict__ in,
                              const float* __restrict__ g,
                              const float* __restrict__ b,
                              float* __restrict__ out, int n) {
    int i = blockIdx.x * blockDim.x + threadIdx.x;
    if (i >= n) return;
    int c = i & 15;
    float v = in[i] * g[c] + b[c];
    out[i] = v > 0.f ? v : 0.f;
}
__global__ void id_add_kernel(const float* __restrict__ u4,
                              const float* __restrict__ h,
                              float* __restrict__ out, int n) {
    int i = blockIdx.x * blockDim.x + threadIdx.x;
    if (i < n) out[i] = u4[i] + h[i];
}
__global__ void nin_add_kernel(const float* __restrict__ u4,
                               const float* __restrict__ h,
                               const float* __restrict__ Ws,
                               float* __restrict__ out, int n) {
    int i = blockIdx.x * blockDim.x + threadIdx.x;
    if (i >= n) return;
    int s = i >> 4, d = i & 15;
    float a = u4[i];
    const float* hr = h + (size_t)s * 16;
    #pragma unroll
    for (int c = 0; c < 16; ++c) a += hr[c] * Ws[c * 16 + d];
    out[i] = a;
}

// ----------------------------------------------------- scatter to dense f16
__global__ void scatter_dense_kernel(const float* __restrict__ h,
                                     const int* __restrict__ px,
                                     const int* __restrict__ py,
                                     const int* __restrict__ pz,
                                     half_t* __restrict__ dense, int Np) {
    int idx = blockIdx.x * blockDim.x + threadIdx.x;
    if (idx >= Np * 16) return;
    int sp = idx >> 4, c = idx & 15;
    dense[(((size_t)px[sp] * 90 + py[sp]) * 90 + pz[sp]) * 16 + c] = (half_t)h[idx];
}
__global__ void mask_scatter_kernel(const int* __restrict__ px,
                                    const int* __restrict__ py,
                                    const int* __restrict__ pz,
                                    unsigned char* __restrict__ m, int Np) {
    int sp = blockIdx.x * blockDim.x + threadIdx.x;
    if (sp < Np) m[((size_t)px[sp] * 90 + py[sp]) * 90 + pz[sp]] = 1;
}
// max over window of 7 along one axis (3-window then 5-window == 7-window)
__global__ void dilate_kernel(const unsigned char* __restrict__ in,
                              unsigned char* __restrict__ out,
                              int n_outer, int a_in, int a_out, int n_inner) {
    int idx = blockIdx.x * blockDim.x + threadIdx.x;
    if (idx >= n_outer * a_out * n_inner) return;
    int i = idx % n_inner;
    int rest = idx / n_inner;
    int a = rest % a_out, o = rest / a_out;
    unsigned char mx = 0;
    #pragma unroll
    for (int w = 0; w < 7; ++w) {
        unsigned char v = in[((size_t)o * a_in + a + w) * n_inner + i];
        mx = v > mx ? v : mx;
    }
    out[idx] = mx;
}

// ------------------------------------------- B-fragment weight swizzle prep
// W layout [co=16][ci=16][ntaps]; Wt layout [npairs][lane=32][e=16] (f16)
// ISA B layout (16x16x32 f16): lane l element e holds B[K=(e<8?0:16)+(l<16?0:8)+(e&7)][N=l&15]
// with K<16 -> tap 2*tp (ci=K), K>=16 -> tap 2*tp+1 (ci=K-16).
__global__ void prep_wt_kernel(const float* __restrict__ W,
                               half_t* __restrict__ Wt,
                               int ntaps, int npairs) {
    int idx = blockIdx.x * blockDim.x + threadIdx.x;
    if (idx >= npairs * 32 * 16) return;
    int e = idx & 15, lane = (idx >> 4) & 31, tp = idx >> 9;
    int K = ((e & 8) ? 16 : 0) + ((lane & 16) ? 8 : 0) + (e & 7);
    int tap = 2 * tp + (K >> 4);
    int ci = K & 15, n = lane & 15;
    half_t v = (half_t)0.f;
    if (tap < ntaps) v = (half_t)W[((size_t)n * 16 + ci) * ntaps + tap];
    Wt[idx] = v;
}

// -------------------------------------------------- dense conv1: 3^3, 90->88
// one wave per 16-z-position tile; D = A(16pos x K) * B(K x 16co), K = 2 taps * 16ci
__global__ __launch_bounds__(128) void conv1_kernel(const half_t* __restrict__ in,
                                                    const half_t* __restrict__ Wt,
                                                    half_t* __restrict__ out) {
    const int lane = threadIdx.x & 31;
    const int wid  = (blockIdx.x << 2) | (threadIdx.x >> 5);
    int zt = wid % 6, rem = wid / 6;
    int y = rem % 88, x = rem / 88;
    if (x >= 88) return;                    // wave-uniform (never taken)
    int z0 = zt * 16; if (z0 > 72) z0 = 72; // overlapping tail tile
    const int m  = lane & 15;
    const int cb = (lane & 16) ? 8 : 0;     // A: K half-group per lane half
    const int rowbase = (((x * 90) + y) * 90 + z0 + m) * 16 + cb;
    v8f acc = {};
    #pragma unroll
    for (int tp = 0; tp < 14; ++tp) {
        const int t0 = 2 * tp, t1 = t0 + 1;
        v16h b = *(const v16h*)(Wt + (((tp << 5) | lane) << 4));
        const int dx0 = t0 / 9, dy0 = (t0 % 9) / 3, dz0 = t0 % 3;
        v8h a0 = *(const v8h*)(in + rowbase + ((dx0 * 8100 + dy0 * 90 + dz0) << 4));
        v8h a1 = {};
        if (t1 < 27) {
            const int dx1 = t1 / 9, dy1 = (t1 % 9) / 3, dz1 = t1 % 3;
            a1 = *(const v8h*)(in + rowbase + ((dx1 * 8100 + dy1 * 90 + dz1) << 4));
        }
        v16h a;
        #pragma unroll
        for (int e = 0; e < 8; ++e) { a[e] = a0[e]; a[e + 8] = a1[e]; }
        acc = __builtin_amdgcn_wmma_f32_16x16x32_f16(false, a, false, b,
                                                     (short)0, acc, false, false);
    }
    const int mb = (lane & 16) ? 8 : 0;     // D: M offset per lane half
    const int n  = lane & 15;
    #pragma unroll
    for (int r = 0; r < 8; ++r) {
        int z = z0 + mb + r;
        out[(((x * 88) + y) * 88 + z) * 16 + n] = (half_t)acc[r];
    }
}

// ------------------- dense conv2: 5^3, 88->84, fused BN+leaky+mask, NCDHW out
__global__ __launch_bounds__(128) void conv2_kernel(const half_t* __restrict__ in,
                                                    const half_t* __restrict__ Wt,
                                                    const float* __restrict__ g_bn,
                                                    const float* __restrict__ b_bn,
                                                    const unsigned char* __restrict__ mask,
                                                    float* __restrict__ dout) {
    const int lane = threadIdx.x & 31;
    const int wid  = (blockIdx.x << 2) | (threadIdx.x >> 5);
    int zt = wid % 6, rem = wid / 6;
    int y = rem % 84, x = rem / 84;
    if (x >= 84) return;
    int z0 = zt * 16; if (z0 > 68) z0 = 68;
    const int m  = lane & 15;
    const int cb = (lane & 16) ? 8 : 0;
    const int rowbase = (((x * 88) + y) * 88 + z0 + m) * 16 + cb;
    v8f acc = {};
    #pragma unroll 3
    for (int tp = 0; tp < 63; ++tp) {
        const int t0 = 2 * tp, t1 = t0 + 1;
        v16h b = *(const v16h*)(Wt + (((tp << 5) | lane) << 4));
        const int dx0 = t0 / 25, dy0 = (t0 % 25) / 5, dz0 = t0 % 5;
        v8h a0 = *(const v8h*)(in + rowbase + ((dx0 * 7744 + dy0 * 88 + dz0) << 4));
        v8h a1 = {};
        if (t1 < 125) {
            const int dx1 = t1 / 25, dy1 = (t1 % 25) / 5, dz1 = t1 % 5;
            a1 = *(const v8h*)(in + rowbase + ((dx1 * 7744 + dy1 * 88 + dz1) << 4));
        }
        v16h a;
        #pragma unroll
        for (int e = 0; e < 8; ++e) { a[e] = a0[e]; a[e + 8] = a1[e]; }
        acc = __builtin_amdgcn_wmma_f32_16x16x32_f16(false, a, false, b,
                                                     (short)0, acc, false, false);
    }
    const int mb = (lane & 16) ? 8 : 0;
    const int n  = lane & 15;
    const float gn = g_bn[n], bn = b_bn[n];
    #pragma unroll
    for (int r = 0; r < 8; ++r) {
        int z = z0 + mb + r;
        float v = acc[r] * gn + bn;
        v = v > 0.f ? v : 0.333f * v;                    // leaky
        v *= (float)mask[((size_t)x * 84 + y) * 84 + z]; // sparse-to-dense mask
        dout[((((size_t)n * 84 + x) * 84) + y) * 84 + z] = v; // NCDHW flatten order
    }
}

// ---------------------------------------------- final 380MB W_lin1 streaming
__global__ __launch_bounds__(256) void dot_reduce_kernel(const float* __restrict__ d,
                                                         const float* __restrict__ W1,
                                                         float* __restrict__ partials,
                                                         int F) {
    __shared__ float sm[256];
    float acc[10];
    #pragma unroll
    for (int j = 0; j < 10; ++j) acc[j] = 0.f;
    for (int i = blockIdx.x * blockDim.x + threadIdx.x; i < F;
         i += gridDim.x * blockDim.x) {
        float v = d[i];
        #pragma unroll
        for (int j = 0; j < 10; ++j) acc[j] += v * W1[(size_t)j * F + i];
    }
    for (int j = 0; j < 10; ++j) {
        sm[threadIdx.x] = acc[j];
        __syncthreads();
        for (int s = 128; s > 0; s >>= 1) {
            if (threadIdx.x < s) sm[threadIdx.x] += sm[threadIdx.x + s];
            __syncthreads();
        }
        if (threadIdx.x == 0) partials[blockIdx.x * 10 + j] = sm[0];
        __syncthreads();
    }
}
__global__ __launch_bounds__(256) void finalize_kernel(const float* __restrict__ partials,
                                                       int nb,
                                                       const float* __restrict__ b_lin1,
                                                       const float* __restrict__ W_lin2,
                                                       const float* __restrict__ b_lin2,
                                                       float* __restrict__ out) {
    __shared__ float sm[256];
    __shared__ float dot[10];
    for (int j = 0; j < 10; ++j) {
        float a = 0.f;
        for (int b = threadIdx.x; b < nb; b += blockDim.x) a += partials[b * 10 + j];
        sm[threadIdx.x] = a;
        __syncthreads();
        for (int s = 128; s > 0; s >>= 1) {
            if (threadIdx.x < s) sm[threadIdx.x] += sm[threadIdx.x + s];
            __syncthreads();
        }
        if (threadIdx.x == 0) dot[j] = sm[0];
        __syncthreads();
    }
    if (threadIdx.x < 2) {
        float o = b_lin2[threadIdx.x];
        for (int j = 0; j < 10; ++j)
            o += W_lin2[threadIdx.x * 10 + j] * (dot[j] + b_lin1[j]);
        out[threadIdx.x] = o;
    }
}

// ===========================================================================
extern "C" void kernel_launch(void* const* d_in, const int* in_sizes, int n_in,
                              void* d_out, int out_size, void* d_ws, size_t ws_size,
                              hipStream_t stream) {
    const float* x_feats = (const float*)d_in[0];
    const float* W1   = (const float*)d_in[1];
    const float* W2   = (const float*)d_in[2];
    const float* W3   = (const float*)d_in[3];
    const float* g1s  = (const float*)d_in[4];
    const float* b1s  = (const float*)d_in[5];
    const float* Wa   = (const float*)d_in[6];
    const float* g2s  = (const float*)d_in[7];
    const float* b2s  = (const float*)d_in[8];
    const float* Wb   = (const float*)d_in[9];
    const float* Ws   = (const float*)d_in[10];
    const float* gf   = (const float*)d_in[11];
    const float* bf   = (const float*)d_in[12];
    const float* Wc1  = (const float*)d_in[13];
    const float* Wc2  = (const float*)d_in[14];
    const float* g_bn = (const float*)d_in[15];
    const float* b_bn = (const float*)d_in[16];
    const float* W_lin1 = (const float*)d_in[17];
    const float* b_lin1 = (const float*)d_in[18];
    const float* W_lin2 = (const float*)d_in[19];
    const float* b_lin2 = (const float*)d_in[20];
    const int* rb1_in = (const int*)d_in[21];
    const int* rb1_out = (const int*)d_in[22];
    const int* rb1_k = (const int*)d_in[23];
    const int* rb2_in = (const int*)d_in[24];
    const int* rb2_out = (const int*)d_in[25];
    const int* rb2_k = (const int*)d_in[26];
    const int* rb3_in = (const int*)d_in[27];
    const int* rb3_out = (const int*)d_in[28];
    const int* rb3_k = (const int*)d_in[29];
    const int* pool_seg = (const int*)d_in[30];
    const int* rb90_in = (const int*)d_in[31];
    const int* rb90_out = (const int*)d_in[32];
    const int* rb90_k = (const int*)d_in[33];
    const int* px = (const int*)d_in[34];
    const int* py = (const int*)d_in[35];
    const int* pz = (const int*)d_in[36];

    const int N   = in_sizes[0];
    const int P1  = in_sizes[21];
    const int P2  = in_sizes[24];
    const int P3  = in_sizes[27];
    const int P90 = in_sizes[31];
    const int Np  = in_sizes[34];
    const int F   = 16 * 84 * 84 * 84;
    const int RB  = 2048; // reduce blocks

    // ---- bump-allocate workspace (256B aligned) ----
    char* base = (char*)d_ws;
    size_t off = 0;
    auto alloc = [&](size_t bytes) -> void* {
        void* p = base + off;
        off = (off + bytes + 255) & ~(size_t)255;
        return p;
    };
    float*    h1      = (float*)alloc((size_t)N * 16 * 4);
    float*    h2      = (float*)alloc((size_t)N * 16 * 4);
    float*    h3      = (float*)alloc((size_t)N * 4 * 4);
    unsigned* hp      = (unsigned*)alloc((size_t)Np * 4 * 4);
    float*    hA      = (float*)alloc((size_t)Np * 16 * 4);
    float*    hB      = (float*)alloc((size_t)Np * 16 * 4);
    float*    u       = (float*)alloc((size_t)Np * 16 * 4);
    float*    u2      = (float*)alloc((size_t)Np * 16 * 4);
    float*    u4      = (float*)alloc((size_t)Np * 16 * 4);
    half_t*   dense90 = (half_t*)alloc((size_t)729000 * 16 * 2);
    half_t*   dense88 = (half_t*)alloc((size_t)681472 * 16 * 2);
    half_t*   Wt1     = (half_t*)alloc((size_t)14 * 512 * 2);
    half_t*   Wt2     = (half_t*)alloc((size_t)63 * 512 * 2);
    unsigned char* mask90 = (unsigned char*)alloc(729000);
    unsigned char* mt1    = (unsigned char*)alloc(680400);
    unsigned char* mt2    = (unsigned char*)alloc(635040);
    unsigned char* mask84 = (unsigned char*)alloc(592704);
    float*    dfinal  = (float*)alloc((size_t)F * 4);
    float*    partials= (float*)alloc((size_t)RB * 10 * 4);

    auto cdiv = [](int a, int b) { return (a + b - 1) / b; };
    auto zero = [&](void* p, size_t bytes) {
        int n = (int)(bytes / 4);
        fill_u32_kernel<<<cdiv(n, 256), 256, 0, stream>>>((unsigned*)p, n, 0u);
    };

    // ---- sparse frontend ----
    zero(h1, (size_t)N * 16 * 4);
    subconv_kernel<<<cdiv(P1 * 16, 256), 256, 0, stream>>>(x_feats, 1, W1, 16,
        rb1_in, rb1_out, rb1_k, P1, h1);
    zero(h2, (size_t)N * 16 * 4);
    subconv_kernel<<<cdiv(P2 * 16, 256), 256, 0, stream>>>(h1, 16, W2, 16,
        rb2_in, rb2_out, rb2_k, P2, h2);
    zero(h3, (size_t)N * 4 * 4);
    subconv_kernel<<<cdiv(P3 * 4, 256), 256, 0, stream>>>(h2, 16, W3, 4,
        rb3_in, rb3_out, rb3_k, P3, h3);

    // ---- max pooling + pad 4->16 ----
    zero(hp, (size_t)Np * 4 * 4);
    pool_max_kernel<<<cdiv(N * 4, 256), 256, 0, stream>>>(h3, pool_seg, hp, N);
    decode_pad_kernel<<<cdiv(Np * 16, 256), 256, 0, stream>>>(hp, hA, Np);

    // ---- 6 ResNet basic blocks ----
    float* hcur = hA;
    float* hnxt = hB;
    const int nin_of[6] = {0, -1, 1, -1, 2, -1};
    for (int i = 0; i < 6; ++i) {
        bnrelu_kernel<<<cdiv(Np * 16, 256), 256, 0, stream>>>(hcur, g1s + i * 16,
            b1s + i * 16, u, Np * 16);
        zero(u2, (size_t)Np * 16 * 4);
        subconv_kernel<<<cdiv(P90 * 16, 256), 256, 0, stream>>>(u, 16,
            Wa + (size_t)i * 27 * 256, 16, rb90_in, rb90_out, rb90_k, P90, u2);
        bnrelu_kernel<<<cdiv(Np * 16, 256), 256, 0, stream>>>(u2, g2s + i * 16,
            b2s + i * 16, u, Np * 16);
        zero(u4, (size_t)Np * 16 * 4);
        subconv_kernel<<<cdiv(P90 * 16, 256), 256, 0, stream>>>(u, 16,
            Wb + (size_t)i * 27 * 256, 16, rb90_in, rb90_out, rb90_k, P90, u4);
        if (nin_of[i] >= 0)
            nin_add_kernel<<<cdiv(Np * 16, 256), 256, 0, stream>>>(u4, hcur,
                Ws + (size_t)nin_of[i] * 256, hnxt, Np * 16);
        else
            id_add_kernel<<<cdiv(Np * 16, 256), 256, 0, stream>>>(u4, hcur, hnxt,
                Np * 16);
        float* t = hcur; hcur = hnxt; hnxt = t;
    }
    bnrelu_kernel<<<cdiv(Np * 16, 256), 256, 0, stream>>>(hcur, gf, bf, hcur, Np * 16);

    // ---- scatter to dense (channel-last f16), weight swizzle ----
    zero(dense90, (size_t)729000 * 16 * 2);
    scatter_dense_kernel<<<cdiv(Np * 16, 256), 256, 0, stream>>>(hcur, px, py, pz,
        dense90, Np);
    prep_wt_kernel<<<cdiv(14 * 512, 256), 256, 0, stream>>>(Wc1, Wt1, 27, 14);
    prep_wt_kernel<<<cdiv(63 * 512, 256), 256, 0, stream>>>(Wc2, Wt2, 125, 63);

    // ---- dilated mask (3-window then 5-window == 7-window, separable) ----
    zero(mask90, 729000);
    mask_scatter_kernel<<<cdiv(Np, 256), 256, 0, stream>>>(px, py, pz, mask90, Np);
    dilate_kernel<<<cdiv(680400, 256), 256, 0, stream>>>(mask90, mt1, 8100, 90, 84, 1);
    dilate_kernel<<<cdiv(635040, 256), 256, 0, stream>>>(mt1, mt2, 90, 90, 84, 84);
    dilate_kernel<<<cdiv(592704, 256), 256, 0, stream>>>(mt2, mask84, 1, 90, 84, 7056);

    // ---- dense WMMA convs ----
    conv1_kernel<<<(88 * 88 * 6) / 4, 128, 0, stream>>>(dense90, Wt1, dense88);
    conv2_kernel<<<(84 * 84 * 6) / 4, 128, 0, stream>>>(dense88, Wt2, g_bn, b_bn,
        mask84, dfinal);

    // ---- final linear layers (streams 380MB W_lin1 once) ----
    dot_reduce_kernel<<<RB, 256, 0, stream>>>(dfinal, W_lin1, partials, F);
    finalize_kernel<<<1, 256, 0, stream>>>(partials, RB, b_lin1, W_lin2, b_lin2,
        (float*)d_out);
}
